// MusicPerformer_67310727462970
// MI455X (gfx1250) — matrix operand order repaired and checked by reading, not compiled, not run on hardware
//
#include <hip/hip_runtime.h>
#include <hip/hip_bf16.h>
#include <math.h>

#define B_SZ    4
#define S_LEN   2048
#define DMODEL  512
#define DFF     2048
#define NHEAD   8
#define HDIM    64
#define NTOK    400
#define NTOKP   448            /* NTOK padded to 64 */
#define NLAYER  12
#define MROWS   (B_SZ * S_LEN) /* 8192 */
#define ATT_EPS 1e-6f

typedef __attribute__((ext_vector_type(16))) __bf16 v16bf;
typedef __attribute__((ext_vector_type(8)))  float  v8f;

// ---------------------------------------------------------------------------
// CDNA5 primitives
// ---------------------------------------------------------------------------
__device__ __forceinline__ v8f wmma_bf16(v16bf a, v16bf b, v8f c) {
  return __builtin_amdgcn_wmma_f32_16x16x32_bf16(
      false, a, false, b, (short)0, c, false, false);
}

// Async global->LDS 16B copy (ASYNCcnt). LDS address = low 32 bits of the
// generic pointer (LDS aperture lives entirely in addr[63:32], ISA 10.2).
__device__ __forceinline__ void async_b128(void* lds, const void* g) {
  const unsigned l = (unsigned)(unsigned long long)lds;
  asm volatile("global_load_async_to_lds_b128 %0, %1, off" :: "v"(l), "v"(g) : "memory");
}
__device__ __forceinline__ void wait_async0() {
  asm volatile("s_wait_asynccnt 0x0" ::: "memory");
}

// A fragment (16x32) from row-major LDS L[row][k]; one contiguous 32B run
// per lane (k = 16*half + s).
__device__ __forceinline__ v16bf lds_frag_A(const __bf16* L, int ld, int row0, int k0) {
  const int lane = threadIdx.x & 31;
  const int m = lane & 15, half = lane >> 4;
  v16bf a;
#pragma unroll
  for (int i = 0; i < 8; ++i) {
    const int kk = k0 + half * 16 + i * 2;
    const __bf16* p = L + (row0 + m) * ld + kk;
    a[2 * i]     = p[0];
    a[2 * i + 1] = p[1];
  }
  return a;
}

// B fragment where logical B[k][n] = L[col0+n][k] (transposed source):
// ONE contiguous 32B run per lane.
__device__ __forceinline__ v16bf lds_frag_BT(const __bf16* L, int ld, int k0, int col0) {
  const int lane = threadIdx.x & 31;
  const int n = lane & 15, half = lane >> 4;
  const __bf16* p = L + (col0 + n) * ld + k0 + half * 16;
  v16bf b;
#pragma unroll
  for (int i = 0; i < 16; ++i) b[i] = p[i];
  return b;
}

// B fragment from the K-pair-swizzled weight tile (flat 32x64 block):
// element (k,n) lives at n*32 + (k/2)*2 + (k&1); 32B run per lane.
__device__ __forceinline__ v16bf lds_frag_Bsw(const __bf16* Bsf, int col0) {
  const int lane = threadIdx.x & 31;
  const int n = col0 + (lane & 15), half = lane >> 4;
  const __bf16* p = Bsf + n * 32 + half * 16;
  v16bf b;
#pragma unroll
  for (int i = 0; i < 16; ++i) b[i] = p[i];
  return b;
}

// ---------------------------------------------------------------------------
// Weight pre-swizzle: f32 [K][N] -> bf16 [K/32][Npad][32] K-pair-packed.
// ---------------------------------------------------------------------------
__global__ __launch_bounds__(256) void swizzle_w_kernel(
    const float* __restrict__ src, __bf16* __restrict__ dst, int K, int N, int Npad) {
  const int idx = blockIdx.x * 256 + threadIdx.x;
  if (idx >= K * Npad) return;
  const int kk = idx / Npad, n = idx % Npad;
  const float v = (n < N) ? src[(size_t)kk * N + n] : 0.0f;
  dst[(size_t)(kk >> 5) * Npad * 32 + (size_t)n * 32 + ((kk & 31) >> 1) * 2 + (kk & 1)] =
      (__bf16)v;
}

// ---------------------------------------------------------------------------
// Embedding * sqrt(d) + sinusoidal PE
// ---------------------------------------------------------------------------
__global__ __launch_bounds__(256) void embed_pe_kernel(
    const int* __restrict__ x, const float* __restrict__ emb, float* __restrict__ h) {
  const int idx = blockIdx.x * 256 + threadIdx.x;
  const int d = idx & (DMODEL - 1);
  const int t = idx >> 9;
  const int s = t & (S_LEN - 1);
  const int tok = x[t];
  const int j2 = d & ~1;
  const float freq = __expf(-(float)j2 * (9.210340371976184f / (float)DMODEL));
  const float ang = (float)s * freq;
  const float pe = (d & 1) ? __cosf(ang) : __sinf(ang);
  h[idx] = emb[tok * DMODEL + d] * 22.62741699796952f + pe;
}

__global__ __launch_bounds__(256) void elu1_kernel(float* __restrict__ p, int n) {
  const int idx = blockIdx.x * 256 + threadIdx.x;
  if (idx < n) {
    const float v = p[idx];
    p[idx] = (v > 0.0f) ? (v + 1.0f) : __expf(v);
  }
}

__global__ __launch_bounds__(128) void lnorm_kernel(
    float* __restrict__ h, const float* __restrict__ g, const float* __restrict__ b) {
  const int row = blockIdx.x, tid = threadIdx.x;
  float* x = h + (size_t)row * DMODEL;
  float s = 0.f, ss = 0.f;
#pragma unroll
  for (int i = tid; i < DMODEL; i += 128) { const float v = x[i]; s += v; ss += v * v; }
  __shared__ float rs[128], rss[128];
  rs[tid] = s; rss[tid] = ss;
  __syncthreads();
  for (int off = 64; off > 0; off >>= 1) {
    if (tid < off) { rs[tid] += rs[tid + off]; rss[tid] += rss[tid + off]; }
    __syncthreads();
  }
  const float mean = rs[0] * (1.0f / DMODEL);
  const float var  = rss[0] * (1.0f / DMODEL) - mean * mean;
  const float rstd = rsqrtf(var + 1e-5f);
#pragma unroll
  for (int i = tid; i < DMODEL; i += 128) x[i] = (x[i] - mean) * rstd * g[i] + b[i];
}

// ---------------------------------------------------------------------------
// Double-buffered WMMA GEMM: C = act(A[M,K]f32 @ Wswz bf16 + bias [+ resid]).
// BM=128, BN=64, BK=64; 8 waves x 2x2 tiles; next tile's B arrives via async
// global->LDS and next A tile rides in registers while WMMAs run.
// ---------------------------------------------------------------------------
#define GBM 128
#define GBN 64
#define GBK 64
#define LDA_P 72   /* 144B rows keep 16B alignment for merged b128 loads */

template <bool RELU>
__global__ __launch_bounds__(256) void gemm_kernel(
    const float* __restrict__ A, const __bf16* __restrict__ Bws,
    const float* __restrict__ bias, const float* __restrict__ resid,
    float* __restrict__ C, int N, int Npad, int K) {
  __shared__ __bf16 As[2][GBM][LDA_P];
  __shared__ alignas(64) __bf16 Bs[2][2][2048];

  const int tid = threadIdx.x;
  const int w = tid >> 5, lane = tid & 31;
  const int wr = w & 3, wc = w >> 2;
  const int n0 = blockIdx.x * GBN;
  const int m0 = blockIdx.y * GBM;
  const int ar = tid >> 4, ac4 = (tid & 15) * 4;   // A-stage coords (row, col4)

  v8f acc[2][2] = {};
  const int ktiles = K / GBK;
  float4 areg[8];

  // ---- prologue: stage tile 0 ----
#pragma unroll
  for (int t2 = 0; t2 < 2; ++t2)
    async_b128(&Bs[0][t2][tid * 8],
               Bws + (size_t)t2 * Npad * 32 + (size_t)n0 * 32 + tid * 8);
#pragma unroll
  for (int i = 0; i < 8; ++i)
    areg[i] = *(const float4*)&A[(size_t)(m0 + ar + i * 16) * K + ac4];
#pragma unroll
  for (int i = 0; i < 8; ++i) {
    const int r = ar + i * 16;
    As[0][r][ac4 + 0] = (__bf16)areg[i].x;
    As[0][r][ac4 + 1] = (__bf16)areg[i].y;
    As[0][r][ac4 + 2] = (__bf16)areg[i].z;
    As[0][r][ac4 + 3] = (__bf16)areg[i].w;
  }
  wait_async0();
  __syncthreads();

  for (int kt = 0; kt < ktiles; ++kt) {
    const int cur = kt & 1, nxt = cur ^ 1;
    const bool more = (kt + 1 < ktiles);
    const int k1 = (kt + 1) * GBK;

    if (more) {
      // issue next B tile async + next A tile global loads (in flight below)
#pragma unroll
      for (int t2 = 0; t2 < 2; ++t2)
        async_b128(&Bs[nxt][t2][tid * 8],
                   Bws + (size_t)((k1 >> 5) + t2) * Npad * 32 + (size_t)n0 * 32 + tid * 8);
#pragma unroll
      for (int i = 0; i < 8; ++i)
        areg[i] = *(const float4*)&A[(size_t)(m0 + ar + i * 16) * K + k1 + ac4];
      if (kt + 2 < ktiles) {  // GL2 prefetch for tile kt+2
        __builtin_prefetch(&A[(size_t)(m0 + ar) * K + k1 + GBK + ac4], 0, 1);
        __builtin_prefetch(
            Bws + (size_t)((k1 >> 5) + 2) * Npad * 32 + (size_t)n0 * 32 + tid * 16, 0, 1);
      }
    }

    // ---- compute current tile: 8 WMMA ----
#pragma unroll
    for (int kh = 0; kh < 2; ++kh) {
      const v16bf a0 = lds_frag_A(&As[cur][0][0], LDA_P, wr * 32, kh * 32);
      const v16bf a1 = lds_frag_A(&As[cur][0][0], LDA_P, wr * 32 + 16, kh * 32);
      const v16bf b0 = lds_frag_Bsw(Bs[cur][kh], wc * 32);
      const v16bf b1 = lds_frag_Bsw(Bs[cur][kh], wc * 32 + 16);
      acc[0][0] = wmma_bf16(a0, b0, acc[0][0]);
      acc[0][1] = wmma_bf16(a0, b1, acc[0][1]);
      acc[1][0] = wmma_bf16(a1, b0, acc[1][0]);
      acc[1][1] = wmma_bf16(a1, b1, acc[1][1]);
    }

    if (more) {
#pragma unroll
      for (int i = 0; i < 8; ++i) {
        const int r = ar + i * 16;
        As[nxt][r][ac4 + 0] = (__bf16)areg[i].x;
        As[nxt][r][ac4 + 1] = (__bf16)areg[i].y;
        As[nxt][r][ac4 + 2] = (__bf16)areg[i].z;
        As[nxt][r][ac4 + 3] = (__bf16)areg[i].w;
      }
      wait_async0();
      __syncthreads();
    }
  }

  // ---- epilogue ----
#pragma unroll
  for (int tm = 0; tm < 2; ++tm) {
#pragma unroll
    for (int tn = 0; tn < 2; ++tn) {
      const int row0 = m0 + wr * 32 + tm * 16 + (lane >> 4) * 8;
      const int col  = n0 + wc * 32 + tn * 16 + (lane & 15);
      if (col < N) {
        const float bv = bias[col];
#pragma unroll
        for (int i = 0; i < 8; ++i) {
          const int row = row0 + i;
          float val = acc[tm][tn][i] + bv;
          if (resid) val += resid[(size_t)row * N + col];
          if (RELU) val = fmaxf(val, 0.0f);
          C[(size_t)row * N + col] = val;
        }
      }
    }
  }
}

// ---------------------------------------------------------------------------
// Chunked causal linear attention, one block per (b,h). Next chunk's Q/K/V
// are prefetched into registers while the current chunk's WMMA stages run.
// ---------------------------------------------------------------------------
#define ACH 64
#define ALD 72

__global__ __launch_bounds__(256) void linattn_kernel(
    const float* __restrict__ q, const float* __restrict__ k,
    const float* __restrict__ v, float* __restrict__ o) {
  __shared__ __bf16 Qc[ACH][ALD], Kc[ACH][ALD];
  __shared__ __bf16 KcT[HDIM][ALD], VcT[HDIM][ALD];   // [dim][pos]
  __shared__ __bf16 Ab[ACH][ALD];                     // masked scores
  __shared__ __bf16 SbT[HDIM][ALD];                   // state^T: SbT[e][d]
  __shared__ float  zst[HDIM], den[ACH];

  const int bh = blockIdx.x;
  const int b = bh >> 3, hh = bh & 7;
  const size_t base = (size_t)b * S_LEN * DMODEL + (size_t)hh * HDIM;
  const int tid = threadIdx.x, w = tid >> 5, lane = tid & 31;
  const int tr = w & 3, tc0 = (w >> 2) * 2;
  const int srow = tid >> 4, sc4 = (tid & 15) * 4;    // staging coords

  for (int i = tid; i < HDIM; i += 256) zst[i] = 0.0f;
  for (int i = tid; i < HDIM * ALD; i += 256) (&SbT[0][0])[i] = (__bf16)0.0f;
  v8f sreg[2] = {};

  float4 qr[4], kr[4], vr[4];
  // preload chunk 0
#pragma unroll
  for (int j = 0; j < 4; ++j) {
    const size_t off = base + (size_t)(srow + j * 16) * DMODEL + sc4;
    qr[j] = *(const float4*)&q[off];
    kr[j] = *(const float4*)&k[off];
    vr[j] = *(const float4*)&v[off];
  }
  __syncthreads();

  for (int c = 0; c < S_LEN / ACH; ++c) {
    // store staged registers -> LDS (row-major + transposed copies)
#pragma unroll
    for (int j = 0; j < 4; ++j) {
      const int r = srow + j * 16;
      const float qv[4] = {qr[j].x, qr[j].y, qr[j].z, qr[j].w};
      const float kv[4] = {kr[j].x, kr[j].y, kr[j].z, kr[j].w};
      const float vv[4] = {vr[j].x, vr[j].y, vr[j].z, vr[j].w};
#pragma unroll
      for (int t = 0; t < 4; ++t) {
        const int d = sc4 + t;
        Qc[r][d]  = (__bf16)qv[t];
        Kc[r][d]  = (__bf16)kv[t];
        KcT[d][r] = (__bf16)kv[t];
        VcT[d][r] = (__bf16)vv[t];
      }
    }
    __syncthreads();

    // issue next chunk's global loads (hidden under WMMA stages)
    if (c + 1 < S_LEN / ACH) {
#pragma unroll
      for (int j = 0; j < 4; ++j) {
        const size_t off =
            base + (size_t)((c + 1) * ACH + srow + j * 16) * DMODEL + sc4;
        qr[j] = *(const float4*)&q[off];
        kr[j] = *(const float4*)&k[off];
        vr[j] = *(const float4*)&v[off];
      }
    }

    // Scores A = Q K^T (B[k=dim][n=key] = Kc[key][dim] -> BT on Kc).
#pragma unroll
    for (int t = 0; t < 2; ++t) {
      const int tcc = tc0 + t;
      v8f a = {};
      a = wmma_bf16(lds_frag_A(&Qc[0][0], ALD, tr * 16, 0),
                    lds_frag_BT(&Kc[0][0], ALD, 0, tcc * 16), a);
      a = wmma_bf16(lds_frag_A(&Qc[0][0], ALD, tr * 16, 32),
                    lds_frag_BT(&Kc[0][0], ALD, 32, tcc * 16), a);
      const int col = tcc * 16 + (lane & 15);
      const int rbase = tr * 16 + (lane >> 4) * 8;
#pragma unroll
      for (int i = 0; i < 8; ++i) {
        const int row = rbase + i;
        Ab[row][col] = (__bf16)((col <= row) ? a[i] : 0.0f);
      }
    }
    __syncthreads();

    // den = eps + rowsum(A) + q . z_prev
    if (tid < ACH) {
      float s = ATT_EPS;
      for (int j = 0; j < ACH; ++j) s += (float)Ab[tid][j];
      for (int d = 0; d < HDIM; ++d) s += (float)Qc[tid][d] * zst[d];
      den[tid] = s;
    }
    __syncthreads();

    // O = Q @ S_prev + A @ V
#pragma unroll
    for (int t = 0; t < 2; ++t) {
      const int tcc = tc0 + t;
      v8f a = {};
      a = wmma_bf16(lds_frag_A(&Qc[0][0], ALD, tr * 16, 0),
                    lds_frag_BT(&SbT[0][0], ALD, 0, tcc * 16), a);
      a = wmma_bf16(lds_frag_A(&Qc[0][0], ALD, tr * 16, 32),
                    lds_frag_BT(&SbT[0][0], ALD, 32, tcc * 16), a);
      a = wmma_bf16(lds_frag_A(&Ab[0][0], ALD, tr * 16, 0),
                    lds_frag_BT(&VcT[0][0], ALD, 0, tcc * 16), a);
      a = wmma_bf16(lds_frag_A(&Ab[0][0], ALD, tr * 16, 32),
                    lds_frag_BT(&VcT[0][0], ALD, 32, tcc * 16), a);
      const int col = tcc * 16 + (lane & 15);
      const int rbase = tr * 16 + (lane >> 4) * 8;
#pragma unroll
      for (int i = 0; i < 8; ++i) {
        const int row = rbase + i;
        o[base + (size_t)(c * ACH + row) * DMODEL + col] = a[i] / den[row];
      }
    }
    __syncthreads();

    // S += K^T @ V : A[m=d][k=pos]=KcT[d][pos], B[k=pos][n=dim]=VcT[dim][pos]
#pragma unroll
    for (int t = 0; t < 2; ++t) {
      const int tcc = tc0 + t;
      sreg[t] = wmma_bf16(lds_frag_A(&KcT[0][0], ALD, tr * 16, 0),
                          lds_frag_BT(&VcT[0][0], ALD, 0, tcc * 16), sreg[t]);
      sreg[t] = wmma_bf16(lds_frag_A(&KcT[0][0], ALD, tr * 16, 32),
                          lds_frag_BT(&VcT[0][0], ALD, 32, tcc * 16), sreg[t]);
      const int col = tcc * 16 + (lane & 15);
      const int rbase = tr * 16 + (lane >> 4) * 8;
#pragma unroll
      for (int i = 0; i < 8; ++i) SbT[col][rbase + i] = (__bf16)sreg[t][i];
    }
    if (tid < HDIM) {
      float s = zst[tid];
      for (int j = 0; j < ACH; ++j) s += (float)KcT[tid][j];
      zst[tid] = s;
    }
    __syncthreads();
  }
}

// ---------------------------------------------------------------------------
// Host-side orchestration
// ---------------------------------------------------------------------------
static inline void launch_gemm(const float* A, const __bf16* Bws, const float* bias,
                               const float* resid, float* C, int N, int Npad, int K,
                               bool relu, hipStream_t stream) {
  dim3 grid(Npad / GBN, MROWS / GBM);
  if (relu)
    gemm_kernel<true><<<grid, 256, 0, stream>>>(A, Bws, bias, resid, C, N, Npad, K);
  else
    gemm_kernel<false><<<grid, 256, 0, stream>>>(A, Bws, bias, resid, C, N, Npad, K);
}

extern "C" void kernel_launch(void* const* d_in, const int* in_sizes, int n_in,
                              void* d_out, int out_size, void* d_ws, size_t ws_size,
                              hipStream_t stream) {
  (void)in_sizes; (void)n_in; (void)out_size; (void)ws_size;
  const int*   x    = (const int*)  d_in[0];
  const float* emb  = (const float*)d_in[1];
  const float* Wqkv = (const float*)d_in[2];
  const float* bqkv = (const float*)d_in[3];
  const float* Wo   = (const float*)d_in[4];
  const float* bo   = (const float*)d_in[5];
  const float* ln1g = (const float*)d_in[6];
  const float* ln1b = (const float*)d_in[7];
  const float* W1   = (const float*)d_in[8];
  const float* b1   = (const float*)d_in[9];
  const float* W2   = (const float*)d_in[10];
  const float* b2   = (const float*)d_in[11];
  const float* ln2g = (const float*)d_in[12];
  const float* ln2b = (const float*)d_in[13];
  const float* lnfg = (const float*)d_in[14];
  const float* lnfb = (const float*)d_in[15];
  const float* Wout = (const float*)d_in[16];
  const float* bout = (const float*)d_in[17];
  float* out = (float*)d_out;

  char* ws = (char*)d_ws;
  const size_t szA = (size_t)MROWS * DMODEL * sizeof(float);
  float* h   = (float*)(ws + 0 * szA);
  float* qb  = (float*)(ws + 1 * szA);
  float* kb  = (float*)(ws + 2 * szA);
  float* vb  = (float*)(ws + 3 * szA);
  float* ab  = (float*)(ws + 4 * szA);
  float* mid = (float*)(ws + 5 * szA);
  char* wsw = ws + 5 * szA + (size_t)MROWS * DFF * sizeof(float);

  const size_t szW512 = (size_t)DMODEL * DMODEL * sizeof(__bf16);
  __bf16* wqkvs = (__bf16*)wsw;
  __bf16* wos   = (__bf16*)(wsw + 36 * szW512);
  __bf16* w1s   = (__bf16*)(wsw + 48 * szW512);
  __bf16* w2s   = (__bf16*)((char*)w1s + 12 * (size_t)DMODEL * DFF * sizeof(__bf16));
  __bf16* wouts = (__bf16*)((char*)w2s + 12 * (size_t)DFF * DMODEL * sizeof(__bf16));

  for (int m = 0; m < 3 * NLAYER; ++m)
    swizzle_w_kernel<<<(DMODEL * DMODEL + 255) / 256, 256, 0, stream>>>(
        Wqkv + (size_t)m * DMODEL * DMODEL, wqkvs + (size_t)m * DMODEL * DMODEL,
        DMODEL, DMODEL, DMODEL);
  for (int l = 0; l < NLAYER; ++l) {
    swizzle_w_kernel<<<(DMODEL * DMODEL + 255) / 256, 256, 0, stream>>>(
        Wo + (size_t)l * DMODEL * DMODEL, wos + (size_t)l * DMODEL * DMODEL,
        DMODEL, DMODEL, DMODEL);
    swizzle_w_kernel<<<(DMODEL * DFF + 255) / 256, 256, 0, stream>>>(
        W1 + (size_t)l * DMODEL * DFF, w1s + (size_t)l * DMODEL * DFF,
        DMODEL, DFF, DFF);
    swizzle_w_kernel<<<(DFF * DMODEL + 255) / 256, 256, 0, stream>>>(
        W2 + (size_t)l * DFF * DMODEL, w2s + (size_t)l * DFF * DMODEL,
        DFF, DMODEL, DMODEL);
  }
  swizzle_w_kernel<<<(DMODEL * NTOKP + 255) / 256, 256, 0, stream>>>(
      Wout, wouts, DMODEL, NTOK, NTOKP);

  const int nElem = MROWS * DMODEL;
  embed_pe_kernel<<<nElem / 256, 256, 0, stream>>>(x, emb, h);

  for (int l = 0; l < NLAYER; ++l) {
    const __bf16* Wq = wqkvs + ((size_t)l * 3 + 0) * DMODEL * DMODEL;
    const __bf16* Wk = wqkvs + ((size_t)l * 3 + 1) * DMODEL * DMODEL;
    const __bf16* Wv = wqkvs + ((size_t)l * 3 + 2) * DMODEL * DMODEL;
    const float* bq = bqkv + ((size_t)l * 3 + 0) * DMODEL;
    const float* bk = bqkv + ((size_t)l * 3 + 1) * DMODEL;
    const float* bv = bqkv + ((size_t)l * 3 + 2) * DMODEL;

    launch_gemm(h, Wq, bq, nullptr, qb, DMODEL, DMODEL, DMODEL, false, stream);
    launch_gemm(h, Wk, bk, nullptr, kb, DMODEL, DMODEL, DMODEL, false, stream);
    launch_gemm(h, Wv, bv, nullptr, vb, DMODEL, DMODEL, DMODEL, false, stream);

    elu1_kernel<<<nElem / 256, 256, 0, stream>>>(qb, nElem);
    elu1_kernel<<<nElem / 256, 256, 0, stream>>>(kb, nElem);

    linattn_kernel<<<B_SZ * NHEAD, 256, 0, stream>>>(qb, kb, vb, ab);

    launch_gemm(ab, wos + (size_t)l * DMODEL * DMODEL, bo + (size_t)l * DMODEL,
                h, h, DMODEL, DMODEL, DMODEL, false, stream);
    lnorm_kernel<<<MROWS, 128, 0, stream>>>(h, ln1g + (size_t)l * DMODEL,
                                            ln1b + (size_t)l * DMODEL);
    launch_gemm(h, w1s + (size_t)l * DMODEL * DFF, b1 + (size_t)l * DFF,
                nullptr, mid, DFF, DFF, DMODEL, true, stream);
    launch_gemm(mid, w2s + (size_t)l * DFF * DMODEL, b2 + (size_t)l * DMODEL,
                h, h, DMODEL, DMODEL, DFF, false, stream);
    lnorm_kernel<<<MROWS, 128, 0, stream>>>(h, ln2g + (size_t)l * DMODEL,
                                            ln2b + (size_t)l * DMODEL);
  }

  lnorm_kernel<<<MROWS, 128, 0, stream>>>(h, lnfg, lnfb);
  launch_gemm(h, wouts, bout, nullptr, out, NTOK, NTOKP, DMODEL, false, stream);
}
